// CombineGraph_75419625718218
// MI455X (gfx1250) — compile-verified, matrix operand order checked by reading, not run on hardware
//
#include <hip/hip_runtime.h>
#include <hip/hip_bf16.h>

// ---------------------------------------------------------------------------
// Fused GCE-GNN LocalAggregator for MI455X (gfx1250, wave32, WMMA).
//   B=512, L=100 (padded to 112 = 7*16), D=128, 4 edge-type attention heads.
// One workgroup (256 threads = 8 waves) per batch element.
// LDS plan (249 KB of 320 KB/WGP):
//   hbf  [112][128] bf16  : h rows                           28,672 B
//   ha [4][112][128] bf16 : h scaled by a_k (WMMA A for e_k) 114,688 B
//   hT  [128][128] bf16   : h transposed (WMMA B for alpha@h) 32,768 B
//   logit[112][112] f32   : selected/masked attention logits  50,176 B
//   alph [112][128] bf16  : softmaxed alpha (WMMA A)          28,672 B
// Matmuls: v_wmma_f32_16x16x32_bf16.  Memory-bound design (~100 MB of HBM
// traffic @ 23.3 TB/s ~ 4.4 us; 6.6 GFLOP is negligible on the WMMA pipes).
// ---------------------------------------------------------------------------

typedef __attribute__((ext_vector_type(16))) __bf16 v16bf;
typedef __attribute__((ext_vector_type(8)))  float  v8f;

#define B_  512
#define L_  100
#define LP  112
#define D_  128
#define NEG_SLOPE 0.2f
#define NEG_INF   (-9e15f)

// LDS carve offsets (bytes)
#define OFF_HBF 0
#define OFF_HA  (OFF_HBF + LP*D_*2)            // 28672
#define OFF_HT  (OFF_HA  + 4*LP*D_*2)          // 143360
#define OFF_LG  (OFF_HT  + D_*D_*2)            // 176128
#define OFF_AL  (OFF_LG  + LP*LP*4)            // 226304
#define SMEM_BYTES (OFF_AL + LP*D_*2)          // 254976

// Load one 16x32 bf16 A/B WMMA fragment half-row from LDS:
// lane holds K elements [0..7] and [16..23] relative to p (two 16B chunks).
__device__ __forceinline__ v16bf frag16(const __bf16* p) {
  v16bf r;
#pragma unroll
  for (int i = 0; i < 8; ++i) { r[i] = p[i]; r[8 + i] = p[16 + i]; }
  return r;
}

__global__ void __launch_bounds__(256)
gce_local_agg(const int* __restrict__ inputs,   // [B,L]
              const int* __restrict__ adj,      // [B,L,L]
              const float* __restrict__ emb,    // [V,D]
              const float* __restrict__ a0,
              const float* __restrict__ a1,
              const float* __restrict__ a2,
              const float* __restrict__ a3,
              float* __restrict__ out)          // [B,L,D]
{
  extern __shared__ __align__(16) char smem[];
  __bf16* hbf   = (__bf16*)(smem + OFF_HBF);   // [LP][D_]
  __bf16* ha    = (__bf16*)(smem + OFF_HA);    // [4][LP][D_]
  __bf16* hT    = (__bf16*)(smem + OFF_HT);    // [D_][D_]
  float*  logit = (float*) (smem + OFF_LG);    // [LP][LP]
  __bf16* alph  = (__bf16*)(smem + OFF_AL);    // [LP][D_]

  const int b    = blockIdx.x;
  const int tid  = threadIdx.x;
  const int lane = tid & 31;
  const int wave = tid >> 5;                   // 0..7
  const int nn    = lane & 15;                 // WMMA N / M-row selector
  const int half8 = (lane >> 4) << 3;          // 0 or 8: K-half selector

  // ---- zero-init hbf + ha + hT (padding rows must be 0 for WMMA) ----
  {
    float4* z = (float4*)smem;
    const int n4 = OFF_LG / 16;                // 176128/16 = 11008
    for (int i = tid; i < n4; i += 256) z[i] = make_float4(0.f, 0.f, 0.f, 0.f);
  }

  // Prefetch the adj slab for this batch element while we gather h.
  {
    const char* ap = (const char*)(adj + (size_t)b * L_ * L_);
    for (int i = tid; i < (L_ * L_ * 4) / 128; i += 256)
      __builtin_prefetch(ap + i * 128, 0, 1);  // -> global_prefetch_b8
  }
  __syncthreads();

  // ---- stage 1: gather h = emb[inputs[b]], build bf16 copies ----
  // Wave w handles rows w, w+8, ... ; each lane moves 4 floats (16B).
  const int d0 = lane * 4;
  const float4 s0 = ((const float4*)a0)[lane];
  const float4 s1 = ((const float4*)a1)[lane];
  const float4 s2 = ((const float4*)a2)[lane];
  const float4 s3 = ((const float4*)a3)[lane];
  const float sc[4][4] = {{s0.x, s0.y, s0.z, s0.w},
                          {s1.x, s1.y, s1.z, s1.w},
                          {s2.x, s2.y, s2.z, s2.w},
                          {s3.x, s3.y, s3.z, s3.w}};
  for (int row = wave; row < L_; row += 8) {
    const int idx = inputs[b * L_ + row];
    const float4 v = ((const float4*)(emb + (size_t)idx * D_))[lane];
    const float vv[4] = {v.x, v.y, v.z, v.w};
#pragma unroll
    for (int i = 0; i < 4; ++i) {
      const __bf16 hb = (__bf16)vv[i];
      hbf[row * D_ + d0 + i]   = hb;
      hT[(d0 + i) * D_ + row]  = hb;
#pragma unroll
      for (int k = 0; k < 4; ++k)
        ha[((k * LP) + row) * D_ + d0 + i] = (__bf16)(vv[i] * sc[k][i]);
    }
  }
  __syncthreads();

  // ---- stage 2: e_k = (h .* a_k) h^T per 16x16 tile; select by adj ----
  // 49 (i,j) tile pairs round-robin over 8 waves; 16 WMMAs per tile pair.
  for (int t = wave; t < 49; t += 8) {
    const int it = t / 7, jt = t % 7;
    const int ib = it * 16, jb = jt * 16;

    // B = h^T : column n is h row (jb+n); contiguous in hbf.
    v16bf bf[4];
#pragma unroll
    for (int kb = 0; kb < 4; ++kb)
      bf[kb] = frag16(&hbf[(jb + nn) * D_ + kb * 32 + half8]);

    v8f e[4];
#pragma unroll
    for (int k = 0; k < 4; ++k) {
      v8f acc = {};
#pragma unroll
      for (int kb = 0; kb < 4; ++kb) {
        const v16bf af = frag16(&ha[((k * LP) + ib + nn) * D_ + kb * 32 + half8]);
        acc = __builtin_amdgcn_wmma_f32_16x16x32_bf16(
            false, af, false, bf[kb], (short)0, acc, false, false);
      }
      e[k] = acc;
    }

    // C/D layout: col N = lane&15, row M = r + (lane>=16 ? 8 : 0).
    const int j = jb + nn;
#pragma unroll
    for (int r = 0; r < 8; ++r) {
      const int i = ib + r + half8;
      int av = 0;
      if (i < L_ && j < L_) av = adj[((size_t)b * L_ + i) * L_ + j];
      float x = NEG_INF;
      if (av >= 1 && av <= 4) {
        const float ek = (av == 1) ? e[0][r]
                       : (av == 2) ? e[1][r]
                       : (av == 3) ? e[2][r]
                       :             e[3][r];
        x = (ek > 0.f) ? ek : NEG_SLOPE * ek;          // leaky_relu
      }
      logit[i * LP + j] = x;
    }
  }
  __syncthreads();

  // ---- stage 3: row softmax over j<100, emit bf16 alpha (K padded to 128) --
  if (tid < LP) {
    const int row = tid;
    float m = NEG_INF;
    for (int j = 0; j < L_; ++j) m = fmaxf(m, logit[row * LP + j]);
    float s = 0.f;
    for (int j = 0; j < L_; ++j) s += __expf(logit[row * LP + j] - m);
    const float inv = 1.f / s;
    for (int j = 0; j < L_; ++j)
      alph[row * D_ + j] = (__bf16)(__expf(logit[row * LP + j] - m) * inv);
    for (int j = L_; j < D_; ++j)
      alph[row * D_ + j] = (__bf16)0.f;
  }
  __syncthreads();

  // ---- stage 4: out = alpha @ h ; B fragments from transposed hT ----
  // 56 (i,d) tile pairs over 8 waves; K = j dim padded to 128 (zeros).
  for (int t = wave; t < 56; t += 8) {
    const int it = t / 8, dt = t % 8;
    const int ib = it * 16, db = dt * 16;
    v8f acc = {};
#pragma unroll
    for (int kb = 0; kb < 4; ++kb) {
      const v16bf af  = frag16(&alph[(ib + nn) * D_ + kb * 32 + half8]);
      const v16bf bfr = frag16(&hT [(db + nn) * D_ + kb * 32 + half8]);
      acc = __builtin_amdgcn_wmma_f32_16x16x32_bf16(
          false, af, false, bfr, (short)0, acc, false, false);
    }
    const int dcol = db + nn;
#pragma unroll
    for (int r = 0; r < 8; ++r) {
      const int i = ib + r + half8;
      if (i < L_)
        out[((size_t)b * L_ + i) * D_ + dcol] = acc[r];
    }
  }
}

extern "C" void kernel_launch(void* const* d_in, const int* in_sizes, int n_in,
                              void* d_out, int out_size, void* d_ws, size_t ws_size,
                              hipStream_t stream) {
  // setup_inputs order: inputs, adj, mask_item, item, emb_table, a0..a3
  const int*   inputs = (const int*)  d_in[0];
  const int*   adj    = (const int*)  d_in[1];
  const float* emb    = (const float*)d_in[4];
  const float* a0     = (const float*)d_in[5];
  const float* a1     = (const float*)d_in[6];
  const float* a2     = (const float*)d_in[7];
  const float* a3     = (const float*)d_in[8];
  float* out = (float*)d_out;

  (void)in_sizes; (void)n_in; (void)out_size; (void)d_ws; (void)ws_size;

  // Opt in to >64KB dynamic LDS (WGP has 320KB); capture-safe host-side call.
  hipFuncSetAttribute((const void*)gce_local_agg,
                      hipFuncAttributeMaxDynamicSharedMemorySize, SMEM_BYTES);

  gce_local_agg<<<B_, 256, SMEM_BYTES, stream>>>(inputs, adj, emb,
                                                 a0, a1, a2, a3, out);
}
